// DeepSpeedMoEBlock_2860448219602
// MI455X (gfx1250) — matrix operand order repaired and checked by reading, not compiled, not run on hardware
//
#include <hip/hip_runtime.h>
#include <hip/hip_bf16.h>
#include <math.h>

// ---------------------------------------------------------------------------
// Types for CDNA5 WMMA (wave32): v_wmma_f32_16x16x32_bf16
// ---------------------------------------------------------------------------
typedef __attribute__((ext_vector_type(16))) __bf16 v16bf;
typedef __attribute__((ext_vector_type(8)))  float  v8f;

union Frag {
    v16bf v;
    uint4 q[2];
};

__device__ __forceinline__ unsigned short f2bf(float f) {
    unsigned int u = __float_as_uint(f);
    unsigned int r = u + 0x7FFFu + ((u >> 16) & 1u);   // round-to-nearest-even
    return (unsigned short)(r >> 16);
}

// ---------------------------------------------------------------------------
// Kernel 1: LayerNorm + gating logits + softmax + top-2 (one block per token)
// ---------------------------------------------------------------------------
__global__ __launch_bounds__(256)
void ln_gate_kernel(const float* __restrict__ x, const float* __restrict__ gamma,
                    const float* __restrict__ beta, const float* __restrict__ wg,
                    float* __restrict__ xn, float* __restrict__ vals,
                    int* __restrict__ idxw, float* __restrict__ gates_ws)
{
    const int t = blockIdx.x;
    const int tid = threadIdx.x;
    const float* xr = x + (size_t)t * 1024;

    __shared__ float red[256];
    __shared__ float lg[8][257];

    float xv[4];
    float s = 0.f;
#pragma unroll
    for (int i = 0; i < 4; ++i) { xv[i] = xr[tid + 256 * i]; s += xv[i]; }

    red[tid] = s; __syncthreads();
    for (int off = 128; off > 0; off >>= 1) {
        if (tid < off) red[tid] += red[tid + off];
        __syncthreads();
    }
    const float mu = red[0] * (1.f / 1024.f);
    __syncthreads();

    float q = 0.f;
#pragma unroll
    for (int i = 0; i < 4; ++i) { float d = xv[i] - mu; q += d * d; }
    red[tid] = q; __syncthreads();
    for (int off = 128; off > 0; off >>= 1) {
        if (tid < off) red[tid] += red[tid + off];
        __syncthreads();
    }
    const float rsig = rsqrtf(red[0] * (1.f / 1024.f) + 1e-5f);
    __syncthreads();

    float lp[8];
#pragma unroll
    for (int e = 0; e < 8; ++e) lp[e] = 0.f;
#pragma unroll
    for (int i = 0; i < 4; ++i) {
        const int h = tid + 256 * i;
        const float v = (xv[i] - mu) * rsig * gamma[h] + beta[h];
        xn[(size_t)t * 1024 + h] = v;
        const float* wr = wg + (size_t)h * 8;
#pragma unroll
        for (int e = 0; e < 8; ++e) lp[e] += v * wr[e];
    }
#pragma unroll
    for (int e = 0; e < 8; ++e) lg[e][tid] = lp[e];
    __syncthreads();

    if (tid < 8) {
        float sum = 0.f;
        for (int i = 0; i < 256; ++i) sum += lg[tid][i];
        lg[tid][256] = sum;                       // logit for expert `tid`
    }
    __syncthreads();

    if (tid == 0) {
        float logit[8], g[8];
        float mx = -1e30f;
        for (int e = 0; e < 8; ++e) { logit[e] = lg[e][256]; mx = fmaxf(mx, logit[e]); }
        float den = 0.f;
        for (int e = 0; e < 8; ++e) { g[e] = expf(logit[e] - mx); den += g[e]; }
        const float inv = 1.f / den;
        int a0 = 0; float b0 = -1.f;
        for (int e = 0; e < 8; ++e) {
            g[e] *= inv;
            gates_ws[(size_t)t * 8 + e] = g[e];
            if (g[e] > b0) { b0 = g[e]; a0 = e; }
        }
        int a1 = 0; float b1 = -1.f;
        for (int e = 0; e < 8; ++e)
            if (e != a0 && g[e] > b1) { b1 = g[e]; a1 = e; }
        vals[2 * t] = b0; vals[2 * t + 1] = b1;
        idxw[2 * t] = a0; idxw[2 * t + 1] = a1;
    }
}

// ---------------------------------------------------------------------------
// Kernel 2: capacity scan. 1 block, 8 waves; wave e owns expert e.
// Reproduces reference order: all tokens slot j=0, then all tokens slot j=1.
// ---------------------------------------------------------------------------
__global__ __launch_bounds__(256)
void scan_kernel(const int* __restrict__ idxw, int* __restrict__ posw,
                 float* __restrict__ counts, int T)
{
    const int tid = threadIdx.x;
    const int wave = tid >> 5;        // expert id
    const int lane = tid & 31;
    int base = 0;
    for (int j = 0; j < 2; ++j) {
        for (int t0 = 0; t0 < T; t0 += 32) {
            const int t = t0 + lane;
            const int e = idxw[2 * t + j];
            const bool m = (e == wave);
            const unsigned mask = __builtin_amdgcn_ballot_w32(m);
            if (m) posw[2 * t + j] = base + __popc(mask & ((1u << lane) - 1u));
            base += __popc(mask);
        }
    }
    if (lane == 0) counts[wave] = (float)base;
}

// ---------------------------------------------------------------------------
// Kernel 3: dispatch kept tokens -> buf[E][C][H] (bf16). One block per (t,j).
// ---------------------------------------------------------------------------
__global__ __launch_bounds__(256)
void dispatch_kernel(const float* __restrict__ xn, const int* __restrict__ idxw,
                     const int* __restrict__ posw, unsigned short* __restrict__ buf,
                     int C)
{
    const int t = blockIdx.x, j = blockIdx.y, tid = threadIdx.x;
    const int p = posw[2 * t + j];
    if (p >= C) return;                       // dropped: combine weight is 0
    const int e = idxw[2 * t + j];
    unsigned short* dst = buf + ((size_t)e * C + p) * 1024;
    const float* src = xn + (size_t)t * 1024;
#pragma unroll
    for (int i = 0; i < 4; ++i) { const int h = tid + 256 * i; dst[h] = f2bf(src[h]); }
}

// ---------------------------------------------------------------------------
// Kernel 4/5: grouped bf16 WMMA GEMM.  D[e] = epilogue(A[e] @ B[e] + bias[e])
//   A: bf16 [M x K] row-major, staged to LDS with GLOBAL_LOAD_ASYNC_TO_LDS_B128
//   B: fp32 [K x N] row-major, converted->bf16 and transposed while staging
//   Block tile: BM=128, BN=128, BK=32 (one WMMA K-slab per stage).
//   8 waves in a 4(M) x 2(N) grid; each wave owns 32x64 = 8 accumulators ->
//   8 v_wmma per stage against 12 ds_load_b128 fragment fetches.
// ---------------------------------------------------------------------------
template <bool GELU_BF16OUT>
__global__ __launch_bounds__(256)
void moe_gemm(const unsigned short* __restrict__ A, const float* __restrict__ B,
              const float* __restrict__ bias, void* __restrict__ Cout,
              int M, int N, int K)
{
    const int e = blockIdx.z;
    const unsigned short* Ae = A + (size_t)e * M * K;
    const float* Be = B + (size_t)e * K * N;
    const float* be = bias + (size_t)e * N;
    const int m0 = blockIdx.x * 128;
    const int n0 = blockIdx.y * 128;

    __shared__ __align__(16) unsigned short As[128][40];  // row-major [m][k], +8 pad
    __shared__ __align__(16) unsigned short Bs[128][40];  // transposed  [n][k], +8 pad

    const int tid = threadIdx.x;
    const int wave = tid >> 5, lane = tid & 31;
    const int hf = lane >> 4, l16 = lane & 15;
    const int wm = (wave >> 1) * 32;   // wave M offset within block tile
    const int wn = (wave & 1) * 64;    // wave N offset within block tile

    // A staging: 128x32 bf16 = 512 b128 chunks; 2 async copies per thread.
    const int rowA0 = tid >> 2;                 // 0..63
    const int colA  = (tid & 3) << 3;           // element offset 0/8/16/24
    const int rowA1 = rowA0 + 64;
    const unsigned ldsA0 = (unsigned)(size_t)&As[rowA0][colA];
    const unsigned ldsA1 = (unsigned)(size_t)&As[rowA1][colA];
    const unsigned gofA0 = (unsigned)(((size_t)(m0 + rowA0) * K + colA) * 2);
    const unsigned gofA1 = (unsigned)(((size_t)(m0 + rowA1) * K + colA) * 2);

    // B staging: 128(n) x 32(k) fp32 -> bf16, transposed into Bs[n][k].
    const int nnB = tid & 127;
    const int kkB = tid >> 7;                   // 0..1

    const v8f vzero = {0.f, 0.f, 0.f, 0.f, 0.f, 0.f, 0.f, 0.f};
    v8f acc[2][4];
#pragma unroll
    for (int mi = 0; mi < 2; ++mi)
#pragma unroll
        for (int ni = 0; ni < 4; ++ni) acc[mi][ni] = vzero;

    for (int k0 = 0; k0 < K; k0 += 32) {
        // --- Stage A via async global->LDS DMA (ASYNCcnt-tracked). ---
        const unsigned short* Asrc = Ae + k0;   // uniform SGPR base
        asm volatile("global_load_async_to_lds_b128 %0, %1, %2 offset:0"
                     :: "v"(ldsA0), "v"(gofA0), "s"(Asrc) : "memory");
        asm volatile("global_load_async_to_lds_b128 %0, %1, %2 offset:0"
                     :: "v"(ldsA1), "v"(gofA1), "s"(Asrc) : "memory");
        // --- Stage B (fp32 -> bf16, transpose) through VGPRs. ---
        for (int kk = kkB; kk < 32; kk += 2)
            Bs[nnB][kk] = f2bf(Be[(size_t)(k0 + kk) * N + n0 + nnB]);
        asm volatile("s_wait_asynccnt 0x0" ::: "memory");
        __syncthreads();

        Frag af[2], bfr[4];
#pragma unroll
        for (int mi = 0; mi < 2; ++mi) {        // A 16-bit layout: lane m=l16,
            const int m = wm + mi * 16 + l16;   // k = hf*8+e (e<8), 16+hf*8+(e-8)
            af[mi].q[0] = *(const uint4*)&As[m][hf * 8];
            af[mi].q[1] = *(const uint4*)&As[m][16 + hf * 8];
        }
#pragma unroll
        for (int ni = 0; ni < 4; ++ni) {        // B 16-bit layout: lane n=l16,
            const int n = wn + ni * 16 + l16;   // k = hf*8+e (e<8), 16+hf*8+(e-8)
            bfr[ni].q[0] = *(const uint4*)&Bs[n][hf * 8];
            bfr[ni].q[1] = *(const uint4*)&Bs[n][16 + hf * 8];
        }
#pragma unroll
        for (int mi = 0; mi < 2; ++mi)
#pragma unroll
            for (int ni = 0; ni < 4; ++ni)
                acc[mi][ni] = __builtin_amdgcn_wmma_f32_16x16x32_bf16(
                    false, af[mi].v, false, bfr[ni].v,
                    (short)0, acc[mi][ni], false, false);
        __syncthreads();
    }

    // Epilogue. C/D layout: lane col = l16, VGPR r -> row = r + 8*hf.
#pragma unroll
    for (int mi = 0; mi < 2; ++mi)
#pragma unroll
        for (int ni = 0; ni < 4; ++ni) {
            const int col = n0 + wn + ni * 16 + l16;
            const float bv = be[col];
#pragma unroll
            for (int r = 0; r < 8; ++r) {
                const int row = m0 + wm + mi * 16 + r + 8 * hf;
                float v = acc[mi][ni][r] + bv;
                if (GELU_BF16OUT) {
                    v = 0.5f * v * (1.0f + erff(v * 0.70710678118654752f));
                    ((unsigned short*)Cout)[(size_t)e * M * N + (size_t)row * N + col] = f2bf(v);
                } else {
                    ((float*)Cout)[(size_t)e * M * N + (size_t)row * N + col] = v;
                }
            }
        }
}

// ---------------------------------------------------------------------------
// Kernel 6: combine + residual. One block per token.
// ---------------------------------------------------------------------------
__global__ __launch_bounds__(256)
void combine_kernel(const float* __restrict__ x, const float* __restrict__ y,
                    const float* __restrict__ vals, const int* __restrict__ idxw,
                    const int* __restrict__ posw, float* __restrict__ out, int C)
{
    const int t = blockIdx.x, tid = threadIdx.x;
    const int i0 = idxw[2 * t], i1 = idxw[2 * t + 1];
    const int p0 = posw[2 * t], p1 = posw[2 * t + 1];
    const float v0 = vals[2 * t], v1 = vals[2 * t + 1];
    const float k0 = (p0 < C) ? v0 : 0.f;
    const float k1 = (p1 < C) ? v1 : 0.f;
    const float inv = 1.f / (k0 + k1 + 1e-9f);
    const float w0 = k0 * inv, w1 = k1 * inv;
    const float* y0 = y + ((size_t)i0 * C + (p0 < C ? p0 : 0)) * 1024;
    const float* y1 = y + ((size_t)i1 * C + (p1 < C ? p1 : 0)) * 1024;
#pragma unroll
    for (int i = 0; i < 4; ++i) {
        const int h = tid + 256 * i;
        float a = x[(size_t)t * 1024 + h];
        if (p0 < C) a += w0 * y0[h];
        if (p1 < C) a += w1 * y1[h];
        out[(size_t)t * 1024 + h] = a;
    }
}

// ---------------------------------------------------------------------------
// Kernel 7: deterministic aux-loss + counts tail.
// ---------------------------------------------------------------------------
__global__ __launch_bounds__(256)
void finalize_kernel(const float* __restrict__ gates, const int* __restrict__ idxw,
                     const float* __restrict__ counts, float* __restrict__ out_tail,
                     int T)
{
    __shared__ float red[256][8];
    __shared__ float meS[8];
    const int tid = threadIdx.x;

    float me[8], ce[8];
#pragma unroll
    for (int e = 0; e < 8; ++e) { me[e] = 0.f; ce[e] = 0.f; }
    for (int t = tid; t < T; t += 256) {
#pragma unroll
        for (int e = 0; e < 8; ++e) me[e] += gates[(size_t)t * 8 + e];
        ce[idxw[2 * t]] += 1.f;
    }

#pragma unroll
    for (int e = 0; e < 8; ++e) red[tid][e] = me[e];
    __syncthreads();
    for (int off = 128; off > 0; off >>= 1) {
        if (tid < off)
            for (int e = 0; e < 8; ++e) red[tid][e] += red[tid + off][e];
        __syncthreads();
    }
    if (tid < 8) meS[tid] = red[0][tid];
    __syncthreads();

#pragma unroll
    for (int e = 0; e < 8; ++e) red[tid][e] = ce[e];
    __syncthreads();
    for (int off = 128; off > 0; off >>= 1) {
        if (tid < off)
            for (int e = 0; e < 8; ++e) red[tid][e] += red[tid + off][e];
        __syncthreads();
    }
    if (tid == 0) {
        const float invT = 1.f / (float)T;
        float s = 0.f;
        for (int e = 0; e < 8; ++e) s += (meS[e] * invT) * (red[0][e] * invT);
        out_tail[0] = s * 8.f;   // mean(me*ce)*E*E == E * sum
    }
    if (tid < 8) out_tail[1 + tid] = counts[tid];
}

// ---------------------------------------------------------------------------
// Host launcher
// ---------------------------------------------------------------------------
extern "C" void kernel_launch(void* const* d_in, const int* in_sizes, int n_in,
                              void* d_out, int out_size, void* d_ws, size_t ws_size,
                              hipStream_t stream) {
    const float* x     = (const float*)d_in[0];
    const float* gamma = (const float*)d_in[1];
    const float* beta  = (const float*)d_in[2];
    const float* wg    = (const float*)d_in[3];
    const float* W1    = (const float*)d_in[4];
    const float* b1    = (const float*)d_in[5];
    const float* W2    = (const float*)d_in[6];
    const float* b2    = (const float*)d_in[7];
    float* out = (float*)d_out;

    const int H = in_sizes[1];              // 1024
    const int E = in_sizes[3] / H;          // 8
    const int F = in_sizes[5] / E;          // 4096
    const int T = in_sizes[0] / H;          // 4096
    const int C = (2 * T + E - 1) / E;      // 1024 (top-2, cap 1.0)

    // Workspace layout (256B-aligned slabs).
    char* ws = (char*)d_ws;
    size_t off = 0;
    auto carve = [&](size_t bytes) -> void* {
        off = (off + 255) & ~(size_t)255;
        void* p = ws + off;
        off += bytes;
        return p;
    };
    float*          xn     = (float*)carve((size_t)T * H * 4);
    float*          vals   = (float*)carve((size_t)T * 2 * 4);
    int*            idxw   = (int*)carve((size_t)T * 2 * 4);
    int*            posw   = (int*)carve((size_t)T * 2 * 4);
    float*          gatesw = (float*)carve((size_t)T * 8 * 4);
    float*          counts = (float*)carve(64);
    unsigned short* buf    = (unsigned short*)carve((size_t)E * C * H * 2);
    unsigned short* h1     = (unsigned short*)carve((size_t)E * C * F * 2);
    float*          ybuf   = (float*)carve((size_t)E * C * H * 4);
    (void)ws_size; (void)n_in; (void)out_size;

    // 1) LayerNorm + gate
    ln_gate_kernel<<<T, 256, 0, stream>>>(x, gamma, beta, wg, xn, vals, idxw, gatesw);
    // 2) capacity scan
    scan_kernel<<<1, 256, 0, stream>>>(idxw, posw, counts, T);
    // 3) zero dispatch buffer, scatter kept tokens (bf16)
    hipMemsetAsync(buf, 0, (size_t)E * C * H * 2, stream);
    dispatch_kernel<<<dim3(T, 2), 256, 0, stream>>>(xn, idxw, posw, buf, C);
    // 4) h1 = gelu(buf @ W1 + b1)  [bf16 out]
    moe_gemm<true><<<dim3(C / 128, F / 128, E), 256, 0, stream>>>(
        buf, W1, b1, (void*)h1, C, F, H);
    // 5) y = h1 @ W2 + b2          [fp32 out]
    moe_gemm<false><<<dim3(C / 128, H / 128, E), 256, 0, stream>>>(
        h1, W2, b2, (void*)ybuf, C, H, F);
    // 6) combine + residual
    combine_kernel<<<T, 256, 0, stream>>>(x, ybuf, vals, idxw, posw, out, C);
    // 7) l_aux + counts tail
    finalize_kernel<<<1, 256, 0, stream>>>(gatesw, idxw, counts,
                                           out + (size_t)T * H, T);
}
